// Attention_3504693314109
// MI455X (gfx1250) — matrix operand order repaired and checked by reading, not compiled, not run on hardware
//
#include <hip/hip_runtime.h>
#include <hip/hip_bf16.h>
#include <math.h>

// Problem constants (from reference)
#define B_N 4
#define S_N 2048
#define D_M 1024
#define H_N 16
#define D_H 64
#define M_TOT (B_N * S_N)   // 8192 flattened tokens

typedef __attribute__((ext_vector_type(16))) __bf16 bf16x16;
typedef __attribute__((ext_vector_type(8)))  __bf16 bf16x8;
typedef __attribute__((ext_vector_type(4)))  __bf16 bf16x4;
typedef __attribute__((ext_vector_type(8)))  float  f32x8;
typedef __attribute__((ext_vector_type(4)))  float  f32x4;
typedef int v4i_ __attribute__((vector_size(16)));  // matches builtin's V4i pointee

// ---------------------------------------------------------------------------
// CDNA5 async global->LDS copy (ASYNCcnt-tracked DMA), with sync fallback
// ---------------------------------------------------------------------------
#if defined(__HIP_DEVICE_COMPILE__) && defined(__gfx1250__) && \
    __has_builtin(__builtin_amdgcn_global_load_async_to_lds_b128)
#define HAS_ASYNC_LDS 1
#else
#define HAS_ASYNC_LDS 0
#endif

static __device__ __forceinline__ void async_cp16(void* lds, const void* g) {
#if HAS_ASYNC_LDS
  // emits: global_load_async_to_lds_b128 (per-lane 16B, tracked by ASYNCcnt)
  __builtin_amdgcn_global_load_async_to_lds_b128(
      (__attribute__((address_space(1))) v4i_*)(unsigned long long)(g),
      (__attribute__((address_space(3))) v4i_*)(unsigned)(unsigned long long)(lds),
      0, 0);
#else
  *(bf16x8*)lds = *(const bf16x8*)g;
#endif
}

static __device__ __forceinline__ void wait_async0() {
#if HAS_ASYNC_LDS
#if __has_builtin(__builtin_amdgcn_s_wait_asynccnt)
  __builtin_amdgcn_s_wait_asynccnt(0);
#else
  asm volatile("s_wait_asynccnt 0x0" ::: "memory");
#endif
#endif
}

// ---------------------------------------------------------------------------
// WMMA wrappers + fragment loaders (layouts per CDNA5 ISA 7.12.2)
// ---------------------------------------------------------------------------
static __device__ __forceinline__ f32x8 wmma_bf16f32(bf16x16 a, bf16x16 b, f32x8 c) {
  return __builtin_amdgcn_wmma_f32_16x16x32_bf16(false, a, false, b, (short)0, c,
                                                 false, false);
}

// A-operand fragment (16x32) from a row-major row pointer.
// Lane holds row (lane&15). elems 0-7 = K[half*8 .. +7], elems 8-15 = K[16+half*8 .. +7]
static __device__ __forceinline__ bf16x16 fragA(const __bf16* rowp, int half) {
  bf16x8 lo = *(const bf16x8*)(rowp + half * 8);
  bf16x8 hi = *(const bf16x8*)(rowp + 16 + half * 8);
  bf16x16 r;
#pragma unroll
  for (int i = 0; i < 8; ++i) { r[i] = lo[i]; r[8 + i] = hi[i]; }
  return r;
}

// B-operand fragment (32x16) from a [n][k] row-major row pointer.
// Lane holds col (lane&15). elems 0-7 = K[half*8 .. +7], elems 8-15 = K[16+half*8 .. +7]
static __device__ __forceinline__ bf16x16 fragB(const __bf16* rowp, int half) {
  bf16x8 lo = *(const bf16x8*)(rowp + half * 8);
  bf16x8 hi = *(const bf16x8*)(rowp + 16 + half * 8);
  bf16x16 r;
#pragma unroll
  for (int i = 0; i < 8; ++i) { r[i] = lo[i]; r[8 + i] = hi[i]; }
  return r;
}

// ---------------------------------------------------------------------------
// Kernel 0: fp32 -> bf16 conversion + transpose (weights)
//   src [k][n] f32  ->  dst [n][k] bf16  (so GEMM B-tiles are DMA-copyable)
// ---------------------------------------------------------------------------
__global__ __launch_bounds__(256) void cvt_transpose_kernel(
    const float* __restrict__ src, __bf16* __restrict__ dst) {
  __shared__ __bf16 tile[64][65];
  const int k0 = blockIdx.x * 64;
  const int n0 = blockIdx.y * 64;
  const int c4 = (threadIdx.x & 15) * 4;
  const int r  = threadIdx.x >> 4;
#pragma unroll
  for (int rr = 0; rr < 4; ++rr) {
    const int kk = rr * 16 + r;
    f32x4 v = *(const f32x4*)(src + (size_t)(k0 + kk) * D_M + n0 + c4);
#pragma unroll
    for (int i = 0; i < 4; ++i) tile[c4 + i][kk] = (__bf16)v[i];
  }
  __syncthreads();
#pragma unroll
  for (int rr = 0; rr < 4; ++rr) {
    const int nn = rr * 16 + r;
    bf16x4 o;
#pragma unroll
    for (int i = 0; i < 4; ++i) o[i] = tile[nn][c4 + i];
    *(bf16x4*)(dst + (size_t)(n0 + nn) * D_M + k0 + c4) = o;
  }
}

// ---------------------------------------------------------------------------
// Tiled WMMA GEMM  Y = X @ W + bias   (W pre-transposed: Wt[n][k])
//   block tile 128x64, BK=32, 8 waves each computing 32x32 (2x2 WMMA tiles)
//   double-buffered LDS, async global->LDS staging
// MODE 0: A fp32,  Out bf16 head layout      [B,H,S,Dh]   (Q,K projections)
// MODE 1: A fp32,  Out bf16 transposed head  [B,H,Dh,S]   (V projection)
// MODE 2: A bf16,  Out fp32 plain            [M, D_M]     (final projection)
// ---------------------------------------------------------------------------
template <int MODE>
__global__ __launch_bounds__(256) void proj_gemm_kernel(
    const void* __restrict__ Aptr, const __bf16* __restrict__ Wt,
    const float* __restrict__ bias, void* __restrict__ Out) {
  const int tid  = threadIdx.x;
  const int wid  = tid >> 5;
  const int lane = tid & 31;
  const int half = lane >> 4;
  const int l16  = lane & 15;
  const int wm   = wid & 3;   // 4 wave rows
  const int wn   = wid >> 2;  // 2 wave cols
  const int row0 = blockIdx.x * 128;
  const int col0 = blockIdx.y * 64;

  constexpr int LDA = 40;  // padded k-stride (bf16)
  constexpr int LDB = 40;
  __shared__ __bf16 As[2][128 * LDA];
  __shared__ __bf16 Bs[2][64 * LDB];  // [n][k]

  f32x8 acc[2][2] = {};

  const int ar  = tid >> 1;          // A stage: row 0..127, 2x16B
  const int ac  = (tid & 1) * 16;
  const int bn  = tid >> 2;          // B stage: n row 0..63, 1x16B
  const int bkg = (tid & 3) * 8;

  auto stage = [&](int buf, int k0) {
    // B tile: 64n x 32k, direct async copies from transposed weights
    async_cp16(&Bs[buf][bn * LDB + bkg],
               Wt + (size_t)(col0 + bn) * D_M + k0 + bkg);
    if (MODE == 2) {
      const __bf16* Af = (const __bf16*)Aptr + (size_t)(row0 + ar) * D_M + k0 + ac;
      async_cp16(&As[buf][ar * LDA + ac], Af);
      async_cp16(&As[buf][ar * LDA + ac + 8], Af + 8);
    } else {
      const float* Af = (const float*)Aptr + (size_t)(row0 + ar) * D_M + k0 + ac;
      f32x4 f0 = *(const f32x4*)(Af + 0);
      f32x4 f1 = *(const f32x4*)(Af + 4);
      f32x4 f2 = *(const f32x4*)(Af + 8);
      f32x4 f3 = *(const f32x4*)(Af + 12);
      bf16x8 b0, b1;
#pragma unroll
      for (int i = 0; i < 4; ++i) {
        b0[i]     = (__bf16)f0[i];
        b0[4 + i] = (__bf16)f1[i];
        b1[i]     = (__bf16)f2[i];
        b1[4 + i] = (__bf16)f3[i];
      }
      *(bf16x8*)&As[buf][ar * LDA + ac]     = b0;
      *(bf16x8*)&As[buf][ar * LDA + ac + 8] = b1;
    }
  };

  stage(0, 0);  // prologue fill

  for (int k0 = 0; k0 < D_M; k0 += 32) {
    const int cur = (k0 >> 5) & 1;
    wait_async0();       // buffer `cur` fully landed (this wave's async ops)
    __syncthreads();     // ...and everyone else's; prior reads of `nxt` done
    if (k0 + 32 < D_M) stage(cur ^ 1, k0 + 32);

#pragma unroll
    for (int mt = 0; mt < 2; ++mt) {
      bf16x16 a = fragA(&As[cur][(wm * 32 + mt * 16 + l16) * LDA], half);
#pragma unroll
      for (int nt = 0; nt < 2; ++nt) {
        bf16x16 bfr = fragB(&Bs[cur][(wn * 32 + nt * 16 + l16) * LDB], half);
        acc[mt][nt] = wmma_bf16f32(a, bfr, acc[mt][nt]);
      }
    }
  }

  // epilogue: bias + layout-specific store
#pragma unroll
  for (int mt = 0; mt < 2; ++mt) {
#pragma unroll
    for (int nt = 0; nt < 2; ++nt) {
#pragma unroll
      for (int j = 0; j < 8; ++j) {
        const int row = row0 + wm * 32 + mt * 16 + j + 8 * half;
        const int col = col0 + wn * 32 + nt * 16 + l16;
        const float v = acc[mt][nt][j] + bias[col];
        if (MODE == 2) {
          ((float*)Out)[(size_t)row * D_M + col] = v;
        } else {
          const int bb = row >> 11;           // / S_N
          const int ss = row & (S_N - 1);
          const int hh = col >> 6;            // / D_H
          const int dd = col & (D_H - 1);
          __bf16* o = (__bf16*)Out;
          if (MODE == 0)
            o[(((size_t)(bb * H_N + hh)) * S_N + ss) * D_H + dd] = (__bf16)v;
          else
            o[(((size_t)(bb * H_N + hh)) * D_H + dd) * S_N + ss] = (__bf16)v;
        }
      }
    }
  }
}

// ---------------------------------------------------------------------------
// Flash attention (non-causal), scale = 1/sqrt(D_MODEL) = 1/32
//   grid = (S/128, B*H); 8 waves; wave owns 16 query rows, loops 64-key blocks
//   double-buffered async K/V staging; V comes in pre-transposed [B,H,Dh,S]
// ---------------------------------------------------------------------------
__global__ __launch_bounds__(256) void flash_attn_kernel(
    const __bf16* __restrict__ Qh, const __bf16* __restrict__ Kh,
    const __bf16* __restrict__ Vt, __bf16* __restrict__ AO) {
  const int tid  = threadIdx.x;
  const int wid  = tid >> 5;
  const int lane = tid & 31;
  const int half = lane >> 4;
  const int l16  = lane & 15;
  const int qb   = blockIdx.x;
  const int bh   = blockIdx.y;
  const int b    = bh >> 4;
  const int h    = bh & 15;

  __shared__ __bf16 Ks[2][64 * 72];    // [key][d], padded
  __shared__ __bf16 Vs[2][64 * 72];    // [dh][key], padded
  __shared__ __bf16 Ps[8][16 * 64];    // per-wave P strip [qrow][key]

  // Q rows for this wave -> register A-fragments (d = 0..63 -> 2 chunks)
  const int qrow0 = qb * 128 + wid * 16 + l16;
  const __bf16* qp = Qh + ((size_t)bh * S_N + qrow0) * D_H;
  const bf16x16 qf0 = fragA(qp, half);
  const bf16x16 qf1 = fragA(qp + 32, half);

  f32x8 o[4] = {};
  float Mx[8], Ls[8];
#pragma unroll
  for (int j = 0; j < 8; ++j) { Mx[j] = -3.0e38f; Ls[j] = 0.f; }

  const float scale = 0.03125f;  // 1/sqrt(1024)

  const int sr = tid >> 2;            // staging row 0..63
  const int sg = (tid & 3) * 16;      // staging col group
  const __bf16* kbase = Kh + ((size_t)bh * S_N + sr) * D_H + sg;
  const __bf16* vbase = Vt + ((size_t)bh * D_H + sr) * S_N + sg;

  auto stageKV = [&](int buf, int kb) {
    const __bf16* kp = kbase + (size_t)kb * D_H;
    const __bf16* vp = vbase + kb;
    async_cp16(&Ks[buf][sr * 72 + sg],     kp);
    async_cp16(&Ks[buf][sr * 72 + sg + 8], kp + 8);
    async_cp16(&Vs[buf][sr * 72 + sg],     vp);
    async_cp16(&Vs[buf][sr * 72 + sg + 8], vp + 8);
    if (kb + 64 < S_N) {  // depth-2: prefetch the tile after the async one
      __builtin_prefetch(kp + 64 * D_H, 0, 0);
      __builtin_prefetch(vp + 64, 0, 0);
    }
  };

  stageKV(0, 0);  // prologue fill

  for (int kb = 0; kb < S_N; kb += 64) {
    const int cur = (kb >> 6) & 1;
    wait_async0();
    __syncthreads();
    if (kb + 64 < S_N) stageKV(cur ^ 1, kb + 64);

    // S = Q K^T for this wave's 16 rows x 64 keys (4 C tiles)
    f32x8 sv[4];
#pragma unroll
    for (int nt = 0; nt < 4; ++nt) {
      const __bf16* krow = &Ks[cur][(nt * 16 + l16) * 72];
      f32x8 c = {};
      c = wmma_bf16f32(qf0, fragB(krow, half), c);
      c = wmma_bf16f32(qf1, fragB(krow + 32, half), c);
      sv[nt] = c;
    }

    // online softmax: row stats via 16-lane shuffles (rows = j + 8*half)
    float mnew[8], alpha[8], rsum[8];
#pragma unroll
    for (int j = 0; j < 8; ++j) {
      float rm = fmaxf(fmaxf(sv[0][j], sv[1][j]), fmaxf(sv[2][j], sv[3][j]));
      rm *= scale;
      rm = fmaxf(rm, __shfl_xor(rm, 1, 32));
      rm = fmaxf(rm, __shfl_xor(rm, 2, 32));
      rm = fmaxf(rm, __shfl_xor(rm, 4, 32));
      rm = fmaxf(rm, __shfl_xor(rm, 8, 32));
      mnew[j]  = fmaxf(Mx[j], rm);
      alpha[j] = __expf(Mx[j] - mnew[j]);
      Mx[j]    = mnew[j];
      rsum[j]  = 0.f;
    }

    // P = exp(scale*S - m), bounced through LDS to re-layout C -> A fragments
#pragma unroll
    for (int nt = 0; nt < 4; ++nt) {
#pragma unroll
      for (int j = 0; j < 8; ++j) {
        const float p = __expf(sv[nt][j] * scale - mnew[j]);
        rsum[j] += p;
        Ps[wid][(j + 8 * half) * 64 + nt * 16 + l16] = (__bf16)p;
      }
    }
#pragma unroll
    for (int j = 0; j < 8; ++j) {
      float rs = rsum[j];
      rs += __shfl_xor(rs, 1, 32);
      rs += __shfl_xor(rs, 2, 32);
      rs += __shfl_xor(rs, 4, 32);
      rs += __shfl_xor(rs, 8, 32);
      Ls[j] = Ls[j] * alpha[j] + rs;
#pragma unroll
      for (int t = 0; t < 4; ++t) o[t][j] *= alpha[j];
    }

    // O += P @ V  (V tile is [dh][key] so B-fragments are contiguous)
    const __bf16* prow = &Ps[wid][l16 * 64];
#pragma unroll
    for (int t = 0; t < 4; ++t) {
      const __bf16* vrow = &Vs[cur][(t * 16 + l16) * 72];
      f32x8 c = o[t];
      c = wmma_bf16f32(fragA(prow, half),      fragB(vrow, half),      c);
      c = wmma_bf16f32(fragA(prow + 32, half), fragB(vrow + 32, half), c);
      o[t] = c;
    }
  }

  // epilogue: normalize and store to [B,S,D_M] bf16 (merged heads)
#pragma unroll
  for (int j = 0; j < 8; ++j) {
    const float inv = 1.0f / Ls[j];
    const int s = qb * 128 + wid * 16 + j + 8 * half;
    __bf16* op = AO + ((size_t)(b * S_N + s)) * D_M + h * D_H;
#pragma unroll
    for (int t = 0; t < 4; ++t) op[t * 16 + l16] = (__bf16)(o[t][j] * inv);
  }
}

// ---------------------------------------------------------------------------
// Host-side orchestration
// ---------------------------------------------------------------------------
extern "C" void kernel_launch(void* const* d_in, const int* in_sizes, int n_in,
                              void* d_out, int out_size, void* d_ws, size_t ws_size,
                              hipStream_t stream) {
  (void)in_sizes; (void)n_in; (void)out_size; (void)ws_size;
  const float* q  = (const float*)d_in[0];
  const float* k  = (const float*)d_in[1];
  const float* v  = (const float*)d_in[2];
  const float* Wq = (const float*)d_in[3];
  const float* bq = (const float*)d_in[4];
  const float* Wk = (const float*)d_in[5];
  const float* bk = (const float*)d_in[6];
  const float* Wv = (const float*)d_in[7];
  const float* bv = (const float*)d_in[8];
  const float* Wo = (const float*)d_in[9];
  const float* bo = (const float*)d_in[10];
  float* out = (float*)d_out;

  // workspace (bf16): 4 transposed weights (2MB ea) + Qh/Kh/Vt/AO (16MB ea) = 72MB
  const size_t WSZ = (size_t)D_M * D_M;                 // 1M elems
  const size_t HSZ = (size_t)B_N * H_N * S_N * D_H;     // 8M elems
  __bf16* Wq_t = (__bf16*)d_ws;
  __bf16* Wk_t = Wq_t + WSZ;
  __bf16* Wv_t = Wk_t + WSZ;
  __bf16* Wo_t = Wv_t + WSZ;
  __bf16* Qh   = Wo_t + WSZ;   // [B,H,S,Dh]
  __bf16* Kh   = Qh + HSZ;     // [B,H,S,Dh]
  __bf16* Vt   = Kh + HSZ;     // [B,H,Dh,S]  (transposed)
  __bf16* AO   = Vt + HSZ;     // [B,S,D_M]

  dim3 tg(16, 16);
  cvt_transpose_kernel<<<tg, 256, 0, stream>>>(Wq, Wq_t);
  cvt_transpose_kernel<<<tg, 256, 0, stream>>>(Wk, Wk_t);
  cvt_transpose_kernel<<<tg, 256, 0, stream>>>(Wv, Wv_t);
  cvt_transpose_kernel<<<tg, 256, 0, stream>>>(Wo, Wo_t);

  dim3 gg(M_TOT / 128, D_M / 64);
  proj_gemm_kernel<0><<<gg, 256, 0, stream>>>(q, Wq_t, bq, Qh);
  proj_gemm_kernel<0><<<gg, 256, 0, stream>>>(k, Wk_t, bk, Kh);
  proj_gemm_kernel<1><<<gg, 256, 0, stream>>>(v, Wv_t, bv, Vt);

  flash_attn_kernel<<<dim3(S_N / 128, B_N * H_N), 256, 0, stream>>>(Qh, Kh, Vt, AO);

  proj_gemm_kernel<2><<<gg, 256, 0, stream>>>(AO, Wo_t, bo, out);
}
